// EGNNLayer_13108240187516
// MI455X (gfx1250) — matrix operand order repaired and checked by reading, not compiled, not run on hardware
//
#include <hip/hip_runtime.h>
#include <hip/hip_bf16.h>
#include <hip/hip_fp16.h>

typedef _Float16 h16;
typedef __attribute__((ext_vector_type(16))) _Float16 v16h;
typedef __attribute__((ext_vector_type(8)))  _Float16 v8h;
typedef __attribute__((ext_vector_type(8)))  float    v8f;

#define WE1_STRIDE 520   // 513 padded to multiple of 8 halves (16B)

union V16U { v16h v; v8h h[2]; };

__device__ __forceinline__ float fast_silu(float x) {
    return x * (1.0f / (1.0f + __expf(-x)));
}

// A-fragment (16x32, 16-bit): lane<16 holds K{k0..k0+7, k0+16..k0+23},
// lane>=16 holds K{k0+8..k0+15, k0+24..k0+31} of its row.
__device__ __forceinline__ v16h load_fragA(const h16* row, int k0, int hi) {
    V16U u;
    u.h[0] = *(const v8h*)(row + k0 + hi * 8);
    u.h[1] = *(const v8h*)(row + k0 + 16 + hi * 8);
    return u.v;
}

// B-fragment (32x16, 16-bit): lane<16 holds K{k0..k0+7, k0+16..k0+23} of its column,
// lane>=16 holds K{k0+8..k0+15, k0+24..k0+31}. Column is contiguous in transposed weights.
__device__ __forceinline__ v16h load_fragB(const h16* col, int k0, int hi) {
    V16U u;
    const h16* p = col + k0 + hi * 8;
    u.h[0] = *(const v8h*)(p);
    u.h[1] = *(const v8h*)(p + 16);
    return u.v;
}

// ---------------------------------------------------------------------------
// Weight prep: f32 [K, 256] row-major  ->  f16 transposed [256, strideT]
// ---------------------------------------------------------------------------
__global__ void prep_weight_kernel(const float* __restrict__ W, h16* __restrict__ Wt,
                                   int K, int Ncols, int strideT) {
    int idx = blockIdx.x * blockDim.x + threadIdx.x;
    if (idx >= K * Ncols) return;
    int k = idx / Ncols;
    int n = idx - k * Ncols;
    Wt[(size_t)n * strideT + k] = (h16)W[idx];
}

// ---------------------------------------------------------------------------
// Edge kernel: 64 edges / 128-thread block, 16 edges per wave.
// e_in(513) -> silu -> silu -> h_e(256); h_e -> silu -> dot(Wc2) -> c
// scatter h_e and dx*c to hagg/xagg with f32 atomics.
// ---------------------------------------------------------------------------
__global__ __launch_bounds__(128) void egnn_edge_kernel(
    const float* __restrict__ feat, const float* __restrict__ coord,
    const int* __restrict__ src, const int* __restrict__ dst,
    const h16* __restrict__ We1t, const float* __restrict__ be1,
    const h16* __restrict__ We2t, const float* __restrict__ be2,
    const h16* __restrict__ Wc1t, const float* __restrict__ bc1,
    const float* __restrict__ Wc2, const float* __restrict__ bc2,
    float* __restrict__ hagg, float* __restrict__ xagg, int E)
{
    __shared__ __align__(16) h16 s_ein[64 * 512];   // gathered [fs|fd] per edge (f16)
    __shared__ __align__(16) h16 s_h1[64 * 256];    // layer-1 out, reused for c-MLP hidden
    __shared__ __align__(16) h16 s_h2[64 * 256];    // h_e (f16)
    __shared__ float s_dx[64][3];
    __shared__ float s_d2[64];
    __shared__ int   s_src[64];
    __shared__ int   s_dst[64];
    __shared__ float s_wc2[256];

    const int t = threadIdx.x;
    const int blockBase = blockIdx.x * 64;

    if (t < 64) {
        int e = blockBase + t; if (e >= E) e = E - 1;
        int s = src[e], d = dst[e];
        s_src[t] = s; s_dst[t] = d;
        float dx0 = coord[3 * s + 0] - coord[3 * d + 0];
        float dx1 = coord[3 * s + 1] - coord[3 * d + 1];
        float dx2 = coord[3 * s + 2] - coord[3 * d + 2];
        s_dx[t][0] = dx0; s_dx[t][1] = dx1; s_dx[t][2] = dx2;
        s_d2[t] = dx0 * dx0 + dx1 * dx1 + dx2 * dx2;
    }
    for (int i = t; i < 256; i += 128) s_wc2[i] = Wc2[i];
    __syncthreads();

    // Stage gathered features (coalesced 512B segments), convert f32 -> f16.
    for (int i = t; i < 64 * 512; i += 128) {
        int eL = i >> 9;
        int k  = i & 511;
        int node = (k < 256) ? s_src[eL] : s_dst[eL];
        s_ein[i] = (h16)feat[(size_t)node * 256 + (k & 255)];
    }
    __syncthreads();

    const int wv = t >> 5, lane = t & 31, lo = lane & 15, hi = lane >> 4;
    const int rowBase = wv * 16;
    const h16* A1 = &s_ein[(rowBase + lo) * 512];

    int   dn[8];
    float d2r[8];
    bool  ev[8];
#pragma unroll
    for (int r = 0; r < 8; ++r) {
        int m = rowBase + r + hi * 8;
        dn[r]  = s_dst[m];
        d2r[r] = s_d2[m];
        ev[r]  = (blockBase + m) < E;
    }

    // ---- layer 1: silu(e_in @ We1 + be1), K=512 via WMMA, d2 row folded in C
#pragma unroll 1
    for (int nt = 0; nt < 16; ++nt) {
        int colN = nt * 16 + lo;
        const h16* B = We1t + (size_t)colN * WE1_STRIDE;
        float bias = be1[colN];
        float w512 = (float)B[512];
        v8f acc;
#pragma unroll
        for (int r = 0; r < 8; ++r) acc[r] = bias + d2r[r] * w512;
#pragma unroll
        for (int kk = 0; kk < 16; ++kk) {
            v16h a = load_fragA(A1, kk * 32, hi);
            v16h b = load_fragB(B, kk * 32, hi);
            acc = __builtin_amdgcn_wmma_f32_16x16x32_f16(false, a, false, b,
                                                         (short)0, acc, false, false);
        }
#pragma unroll
        for (int r = 0; r < 8; ++r) {
            float sv = fast_silu(acc[r]);
            s_h1[(rowBase + r + hi * 8) * 256 + colN] = (h16)sv;
        }
    }

    // ---- layer 2: h_e = silu(h1 @ We2 + be2); scatter h_e to hagg
    const h16* A2 = &s_h1[(rowBase + lo) * 256];
#pragma unroll 1
    for (int nt = 0; nt < 16; ++nt) {
        int colN = nt * 16 + lo;
        const h16* B = We2t + (size_t)colN * 256;
        float bias = be2[colN];
        v8f acc;
#pragma unroll
        for (int r = 0; r < 8; ++r) acc[r] = bias;
#pragma unroll
        for (int kk = 0; kk < 8; ++kk) {
            v16h a = load_fragA(A2, kk * 32, hi);
            v16h b = load_fragB(B, kk * 32, hi);
            acc = __builtin_amdgcn_wmma_f32_16x16x32_f16(false, a, false, b,
                                                         (short)0, acc, false, false);
        }
#pragma unroll
        for (int r = 0; r < 8; ++r) {
            float sv = fast_silu(acc[r]);
            s_h2[(rowBase + r + hi * 8) * 256 + colN] = (h16)sv;
            if (ev[r]) unsafeAtomicAdd(&hagg[(size_t)dn[r] * 256 + colN], sv);
        }
    }

    // ---- layer 3: tmid = silu(h_e @ Wc1 + bc1)  (into s_h1, reused)
    const h16* A3 = &s_h2[(rowBase + lo) * 256];
#pragma unroll 1
    for (int nt = 0; nt < 16; ++nt) {
        int colN = nt * 16 + lo;
        const h16* B = Wc1t + (size_t)colN * 256;
        float bias = bc1[colN];
        v8f acc;
#pragma unroll
        for (int r = 0; r < 8; ++r) acc[r] = bias;
#pragma unroll
        for (int kk = 0; kk < 8; ++kk) {
            v16h a = load_fragA(A3, kk * 32, hi);
            v16h b = load_fragB(B, kk * 32, hi);
            acc = __builtin_amdgcn_wmma_f32_16x16x32_f16(false, a, false, b,
                                                         (short)0, acc, false, false);
        }
#pragma unroll
        for (int r = 0; r < 8; ++r) {
            s_h1[(rowBase + r + hi * 8) * 256 + colN] = (h16)fast_silu(acc[r]);
        }
    }

    // ---- c = tmid @ Wc2 + bc2 ; x_e = dx * c -> scatter to xagg
    if (hi == 0) {
        int eL = rowBase + lo;
        if (blockBase + eL < E) {
            const h16* tr = &s_h1[eL * 256];
            float sum = bc2[0];
#pragma unroll
            for (int k = 0; k < 256; k += 8) {
                v8h tv = *(const v8h*)(tr + k);
#pragma unroll
                for (int j = 0; j < 8; ++j) sum += (float)tv[j] * s_wc2[k + j];
            }
            int d = s_dst[eL];
            unsafeAtomicAdd(&xagg[d * 3 + 0], s_dx[eL][0] * sum);
            unsafeAtomicAdd(&xagg[d * 3 + 1], s_dx[eL][1] * sum);
            unsafeAtomicAdd(&xagg[d * 3 + 2], s_dx[eL][2] * sum);
        }
    }
}

// ---------------------------------------------------------------------------
// Node kernel: 64 nodes / block; h = silu([feat, h_agg] @ Wn1 + bn1) @ Wn2 + bn2
// x = coordinate + x_agg
// ---------------------------------------------------------------------------
__global__ __launch_bounds__(128) void egnn_node_kernel(
    const float* __restrict__ feat, const float* __restrict__ coord,
    const float* __restrict__ hagg, const float* __restrict__ xagg,
    const h16* __restrict__ Wn1t, const float* __restrict__ bn1,
    const h16* __restrict__ Wn2t, const float* __restrict__ bn2,
    float* __restrict__ out_h, float* __restrict__ out_x, int N)
{
    __shared__ __align__(16) h16 s_nin[64 * 512];
    __shared__ __align__(16) h16 s_hn[64 * 256];

    const int t = threadIdx.x;
    const int blockBase = blockIdx.x * 64;

    for (int i = t; i < 64 * 512; i += 128) {
        int nL = i >> 9;
        int k  = i & 511;
        int node = blockBase + nL; if (node >= N) node = N - 1;
        float v = (k < 256) ? feat[(size_t)node * 256 + k]
                            : hagg[(size_t)node * 256 + (k - 256)];
        s_nin[i] = (h16)v;
    }
    __syncthreads();

    const int wv = t >> 5, lane = t & 31, lo = lane & 15, hi = lane >> 4;
    const int rowBase = wv * 16;
    const h16* A1 = &s_nin[(rowBase + lo) * 512];

    // ---- layer 1: silu(n_in @ Wn1 + bn1)
#pragma unroll 1
    for (int nt = 0; nt < 16; ++nt) {
        int colN = nt * 16 + lo;
        const h16* B = Wn1t + (size_t)colN * 512;
        float bias = bn1[colN];
        v8f acc;
#pragma unroll
        for (int r = 0; r < 8; ++r) acc[r] = bias;
#pragma unroll
        for (int kk = 0; kk < 16; ++kk) {
            v16h a = load_fragA(A1, kk * 32, hi);
            v16h b = load_fragB(B, kk * 32, hi);
            acc = __builtin_amdgcn_wmma_f32_16x16x32_f16(false, a, false, b,
                                                         (short)0, acc, false, false);
        }
#pragma unroll
        for (int r = 0; r < 8; ++r) {
            s_hn[(rowBase + r + hi * 8) * 256 + colN] = (h16)fast_silu(acc[r]);
        }
    }

    // ---- layer 2: out_h = hn @ Wn2 + bn2 (no activation), stores guarded
    const h16* A2 = &s_hn[(rowBase + lo) * 256];
#pragma unroll 1
    for (int nt = 0; nt < 16; ++nt) {
        int colN = nt * 16 + lo;
        const h16* B = Wn2t + (size_t)colN * 256;
        float bias = bn2[colN];
        v8f acc;
#pragma unroll
        for (int r = 0; r < 8; ++r) acc[r] = bias;
#pragma unroll
        for (int kk = 0; kk < 8; ++kk) {
            v16h a = load_fragA(A2, kk * 32, hi);
            v16h b = load_fragB(B, kk * 32, hi);
            acc = __builtin_amdgcn_wmma_f32_16x16x32_f16(false, a, false, b,
                                                         (short)0, acc, false, false);
        }
#pragma unroll
        for (int r = 0; r < 8; ++r) {
            int node = blockBase + rowBase + r + hi * 8;
            if (node < N) out_h[(size_t)node * 256 + colN] = acc[r];
        }
    }

    // ---- x = coordinate + x_agg
    if (t < 64) {
        int node = blockBase + t;
        if (node < N) {
#pragma unroll
            for (int c = 0; c < 3; ++c)
                out_x[node * 3 + c] = coord[node * 3 + c] + xagg[node * 3 + c];
        }
    }
}

// ---------------------------------------------------------------------------
extern "C" void kernel_launch(void* const* d_in, const int* in_sizes, int n_in,
                              void* d_out, int out_size, void* d_ws, size_t ws_size,
                              hipStream_t stream) {
    const float* feat  = (const float*)d_in[0];
    const float* coord = (const float*)d_in[1];
    const int*   src   = (const int*)d_in[2];
    const int*   dst   = (const int*)d_in[3];
    const float* We1 = (const float*)d_in[4];  const float* be1 = (const float*)d_in[5];
    const float* We2 = (const float*)d_in[6];  const float* be2 = (const float*)d_in[7];
    const float* Wc1 = (const float*)d_in[8];  const float* bc1 = (const float*)d_in[9];
    const float* Wc2 = (const float*)d_in[10]; const float* bc2 = (const float*)d_in[11];
    const float* Wn1 = (const float*)d_in[12]; const float* bn1 = (const float*)d_in[13];
    const float* Wn2 = (const float*)d_in[14]; const float* bn2 = (const float*)d_in[15];

    const int N = in_sizes[0] / 256;
    const int E = in_sizes[2];

    // Workspace layout
    char* ws = (char*)d_ws;
    float* hagg = (float*)ws;
    size_t off = (size_t)N * 256 * sizeof(float);
    float* xagg = (float*)(ws + off);
    off += (size_t)N * 3 * sizeof(float);
    size_t zeroBytes = off;
    off = (off + 255) & ~(size_t)255;
    h16* We1t = (h16*)(ws + off); off += (size_t)256 * WE1_STRIDE * 2;
    h16* We2t = (h16*)(ws + off); off += (size_t)256 * 256 * 2;
    h16* Wc1t = (h16*)(ws + off); off += (size_t)256 * 256 * 2;
    h16* Wn1t = (h16*)(ws + off); off += (size_t)256 * 512 * 2;
    h16* Wn2t = (h16*)(ws + off); off += (size_t)256 * 256 * 2;

    hipMemsetAsync(d_ws, 0, zeroBytes, stream);

    // Transpose + convert weights to f16 (deterministic, every launch)
    prep_weight_kernel<<<(513 * 256 + 255) / 256, 256, 0, stream>>>(We1, We1t, 513, 256, WE1_STRIDE);
    prep_weight_kernel<<<(256 * 256 + 255) / 256, 256, 0, stream>>>(We2, We2t, 256, 256, 256);
    prep_weight_kernel<<<(256 * 256 + 255) / 256, 256, 0, stream>>>(Wc1, Wc1t, 256, 256, 256);
    prep_weight_kernel<<<(512 * 256 + 255) / 256, 256, 0, stream>>>(Wn1, Wn1t, 512, 256, 512);
    prep_weight_kernel<<<(256 * 256 + 255) / 256, 256, 0, stream>>>(Wn2, Wn2t, 256, 256, 256);

    egnn_edge_kernel<<<(E + 63) / 64, 128, 0, stream>>>(
        feat, coord, src, dst, We1t, be1, We2t, be2, Wc1t, bc1, Wc2, bc2,
        hagg, xagg, E);

    float* out_h = (float*)d_out;
    float* out_x = out_h + (size_t)N * 256;
    egnn_node_kernel<<<(N + 63) / 64, 128, 0, stream>>>(
        feat, coord, hagg, xagg, Wn1t, bn1, Wn2t, bn2, out_h, out_x, N);
}
